// XNORPlusPlusConv1d_75170517614985
// MI455X (gfx1250) — 8-bit matrix operand order repaired and checked by reading, compiled, not run on hardware
//
#include <hip/hip_runtime.h>

typedef int   int4v __attribute__((ext_vector_type(4)));
typedef int   v8i   __attribute__((ext_vector_type(8)));
typedef int   v16i  __attribute__((ext_vector_type(16)));
typedef float v8f   __attribute__((ext_vector_type(8)));

static __device__ inline v8i cat4(int4v a, int4v b) {
    return __builtin_shufflevector(a, b, 0,1,2,3,4,5,6,7);
}
static __device__ inline v16i cat8i(v8i a, v8i b) {
    return __builtin_shufflevector(a, b, 0,1,2,3,4,5,6,7,8,9,10,11,12,13,14,15);
}

static __device__ inline unsigned char bsign_fp8(float v) {
    return (v > 0.0f) ? 0x38u : ((v < 0.0f) ? 0xB8u : 0x00u);
}

#define L_LEN   2000
#define C_IN    32
#define C_OUT   32
#define KW      3
#define TILE_L  256
#define XCOLS   (TILE_L + 2)
#define XSTR    48
#define WROW    80

__global__ __launch_bounds__(128)
void xnor_conv1d_wmma_fp8(const float* __restrict__ x, const float* __restrict__ w,
                          const float* __restrict__ osc, const float* __restrict__ lsc,
                          float* __restrict__ y)
{
    __shared__ unsigned char xlds[XCOLS * XSTR];
    __shared__ unsigned char wshuf[4 * 16 * WROW];

    const int tid = threadIdx.x;
    const int n   = blockIdx.y;
    const int lb  = blockIdx.x * TILE_L;

    for (int i = tid; i < C_IN * XCOLS; i += 128) {
        int ci = i / XCOLS;
        int c  = i - ci * XCOLS;
        int l  = lb - 1 + c;
        unsigned char v = 0;
        if (l >= 0 && l < L_LEN)
            v = bsign_fp8(x[(n * C_IN + ci) * L_LEN + l]);
        xlds[c * XSTR + ci] = v;
    }
    for (int i = tid; i < 4 * 16 * 64; i += 128) {
        int b   = i & 3;
        int v   = (i >> 2) & 15;
        int row = (i >> 6) & 15;
        int gm  = i >> 10;
        int g   = gm & 1;
        int mt  = gm >> 1;
        int K   = 32 * (v >> 2) + 16 * g + 4 * (v & 3) + b;
        int k   = K >> 5;
        int ci  = K & 31;
        unsigned char val = 0;
        if (k < KW) {
            int co = mt * 16 + row;
            val = bsign_fp8(w[(co * C_IN + ci) * KW + k]);
        }
        wshuf[gm * (16 * WROW) + row * WROW + (i & 63)] = val;
    }
    __syncthreads();

    const int lane = tid & 31;
    const int wv   = tid >> 5;
    const int row  = lane & 15;
    const int g    = lane >> 4;

    v16i a[2];
#pragma unroll
    for (int mt = 0; mt < 2; ++mt) {
        const unsigned char* base = &wshuf[((mt * 2 + g) * 16 + row) * WROW];
        int4v q0 = *(const int4v*)(base + 0);
        int4v q1 = *(const int4v*)(base + 16);
        int4v q2 = *(const int4v*)(base + 32);
        int4v q3 = *(const int4v*)(base + 48);
        a[mt] = cat8i(cat4(q0, q1), cat4(q2, q3));
    }

    float os0[8], os1[8];
#pragma unroll
    for (int v = 0; v < 8; ++v) {
        os0[v] = osc[v + 8 * g];
        os1[v] = osc[16 + v + 8 * g];
    }

    const int4v zero4 = {0, 0, 0, 0};

    for (int s = 0; s < 4; ++s) {
        const int off = wv * 64 + s * 16 + row;
        int4v c0 = *(const int4v*)&xlds[(off + 0) * XSTR + g * 16];
        int4v c1 = *(const int4v*)&xlds[(off + 1) * XSTR + g * 16];
        int4v c2 = *(const int4v*)&xlds[(off + 2) * XSTR + g * 16];
        v16i bf = cat8i(cat4(c0, c1), cat4(c2, zero4));

        v8f acc0 = {}, acc1 = {};
        acc0 = __builtin_amdgcn_wmma_f32_16x16x128_fp8_fp8(a[0], bf, (short)0, acc0, false, false);
        acc1 = __builtin_amdgcn_wmma_f32_16x16x128_fp8_fp8(a[1], bf, (short)0, acc1, false, false);

        const int pos = lb + off;
        if (pos < L_LEN) {
            const float ls = lsc[pos];
#pragma unroll
            for (int v = 0; v < 8; ++v) {
                const int co = v + 8 * g;
                y[(n * C_OUT + co) * L_LEN + pos]      = acc0[v] * os0[v] * ls;
                y[(n * C_OUT + co + 16) * L_LEN + pos] = acc1[v] * os1[v] * ls;
            }
        }
    }
}

extern "C" void kernel_launch(void* const* d_in, const int* in_sizes, int n_in,
                              void* d_out, int out_size, void* d_ws, size_t ws_size,
                              hipStream_t stream) {
    const float* x   = (const float*)d_in[0];
    const float* w   = (const float*)d_in[1];
    const float* osc = (const float*)d_in[2];
    const float* lsc = (const float*)d_in[3];
    float* y = (float*)d_out;

    const int N = in_sizes[0] / (C_IN * L_LEN);
    dim3 grid((L_LEN + TILE_L - 1) / TILE_L, N);
    hipLaunchKernelGGL(xnor_conv1d_wmma_fp8, grid, dim3(128), 0, stream, x, w, osc, lsc, y);
}
